// RoIHeads_52458730554160
// MI455X (gfx1250) — compile-verified, matrix operand order checked by reading, not run on hardware
//
#include <hip/hip_runtime.h>
#include <cstdint>

// ---------------------------------------------------------------------------
// RoIHeads postprocess for MI455X (gfx1250, wave32).
//
// Pipeline (all on `stream`, all scratch in d_ws):
//   k_init     : zero per-image max-coord accumulators
//   k_scores   : softmax (WMMA row-sum) + box decode + mask -> masked scores
//   k_topk     : histogram threshold + LDS compaction + bitonic sort -> top2048
//   k_decode   : decode the 2048 selected boxes, per-image max-coord
//   k_offset   : batched-NMS per-label coordinate offset
//   k_nmsmask  : TDM tensor_load_to_lds of 32KB box tile; 32-bit IoU bitmasks
//   k_nms_scan : serial greedy scan (1 wave / image) -> top-100 dets + labels
//
// CDNA5-specific: v_wmma_f32_16x16x32_f16 (softmax denominators),
// tensor_load_to_lds + s_wait_tensorcnt (TDM staging of NMS box tile).
// ---------------------------------------------------------------------------

typedef __attribute__((ext_vector_type(16))) _Float16 v16h;
typedef __attribute__((ext_vector_type(8)))  float    v8f;
typedef __attribute__((ext_vector_type(4)))  unsigned v4u;
typedef __attribute__((ext_vector_type(8)))  int      v8i;
typedef __attribute__((ext_vector_type(4)))  int      v4i;

#define B_IMG   8
#define N_PROP  2000
#define C_CLS   91
#define NFG     90                 // C-1 foreground classes
#define M_CAND  (N_PROP * NFG)     // 180000 candidates per image
#define KPRE    2048
#define DETS    100
#define CAP     4096               // compaction capacity before bitonic sort
#define SCORE_T 0.05f
#define NMS_T   0.5f
#define IMG_WH  800.0f
#define XCLIP   4.135166556742356f
#define NEG_BIG -1e30f

// ---------------------------------------------------------------------------
// Kernel A: per 16-row tile, one wave. Softmax denominators via WMMA
// (exp-matrix [16x32] x ones [32x16] -> replicated row sums in f32 acc),
// then per-candidate decode + masking. Lane pair (L, L^16) shares row L&15.
// ---------------------------------------------------------------------------
__global__ __launch_bounds__(32)
void k_scores(const float* __restrict__ logits, const float* __restrict__ deltas,
              const float* __restrict__ props,  float* __restrict__ scores) {
  const int img  = blockIdx.x / (N_PROP / 16);
  const int tile = blockIdx.x % (N_PROP / 16);
  const int lane = threadIdx.x;
  const int r    = lane & 15;
  const int n    = tile * 16 + r;
  const bool hi  = (lane & 16) != 0;

  const float* lrow = logits + (size_t)(img * N_PROP + n) * C_CLS;

  // Row max (for softmax stability): halves combined across the lane pair.
  float mx = -__builtin_inff();
  const int mc0 = hi ? 46 : 0, mc1 = hi ? 91 : 46;
  for (int c = mc0; c < mc1; ++c) mx = fmaxf(mx, lrow[c]);
  mx = fmaxf(mx, __shfl_xor(mx, 16, 32));

  // WMMA denominator: B = ones, so only "each K covered exactly once per
  // lane pair" matters, not the exact K slot order. 3 chunks cover 96 >= 91.
  v16h ones;
#pragma unroll
  for (int e = 0; e < 16; ++e) ones[e] = (_Float16)1.0f;

  v8f acc = {};
#pragma unroll
  for (int ch = 0; ch < 3; ++ch) {
    const int base = ch * 32 + (hi ? 16 : 0);
    v16h a;
#pragma unroll
    for (int e = 0; e < 16; ++e) {
      const int c = base + e;
      // Branch-free padding: clamp the (divergent) index so the load is
      // unconditional, then cndmask the pad lanes to 0 after the exp.
      const int cc = (c < C_CLS) ? c : 0;
      float ev = __expf(lrow[cc] - mx);
      ev = (c < C_CLS) ? ev : 0.0f;
      a[e] = (_Float16)ev;
    }
    acc = __builtin_amdgcn_wmma_f32_16x16x32_f16(
        /*neg_a=*/false, a, /*neg_b=*/false, ones,
        /*c_mod=*/(short)0, acc, /*reuse_a=*/false, /*reuse_b=*/false);
  }

  // D layout: VGPR g of lane L holds row (g + 8*(L>=16)); columns identical.
  const int g = lane & 7;
  float dsel = (g == 0) ? acc[0] : (g == 1) ? acc[1] : (g == 2) ? acc[2] :
               (g == 3) ? acc[3] : (g == 4) ? acc[4] : (g == 5) ? acc[5] :
               (g == 6) ? acc[6] : acc[7];
  const float dOther = __shfl_xor(dsel, 16, 32);
  const bool here    = (((lane & 8) == 0) == ((lane & 16) == 0));
  const float denom  = here ? dsel : dOther;

  // Proposal geometry for row n.
  const float* pr = props + (size_t)(img * N_PROP + n) * 4;
  const float px1 = pr[0], py1 = pr[1], px2 = pr[2], py2 = pr[3];
  const float pw = px2 - px1, ph = py2 - py1;
  const float pcx = px1 + 0.5f * pw, pcy = py1 + 0.5f * ph;

  const float* drow = deltas + (size_t)(img * N_PROP + n) * C_CLS * 4;
  float* srow = scores + (size_t)img * M_CAND + (size_t)n * NFG;

  // 90 foreground classes split 45/45 across the lane pair.
  const int cs = hi ? 46 : 1, ce = hi ? 91 : 46;
  for (int c = cs; c < ce; ++c) {
    const float sc = __expf(lrow[c] - mx) / denom;
    const float dx = drow[4 * c + 0] * 0.1f;               // /WX
    const float dy = drow[4 * c + 1] * 0.1f;               // /WY
    const float dw = fminf(drow[4 * c + 2] * 0.2f, XCLIP); // /WW, clipped
    const float dh = fminf(drow[4 * c + 3] * 0.2f, XCLIP);
    const float cx = dx * pw + pcx, cy = dy * ph + pcy;
    const float bw = __expf(dw) * pw, bh = __expf(dh) * ph;
    const float x1 = fminf(fmaxf(cx - 0.5f * bw, 0.f), IMG_WH);
    const float y1 = fminf(fmaxf(cy - 0.5f * bh, 0.f), IMG_WH);
    const float x2 = fminf(fmaxf(cx + 0.5f * bw, 0.f), IMG_WH);
    const float y2 = fminf(fmaxf(cy + 0.5f * bh, 0.f), IMG_WH);
    const bool ok = (sc > SCORE_T) && ((x2 - x1) >= 1.f) && ((y2 - y1) >= 1.f);
    srow[c - 1] = ok ? sc : -__builtin_inff();
  }
}

// ---------------------------------------------------------------------------
// Kernel B: per-image top-2048 (sorted desc). 256-bin histogram -> threshold
// bin tb with suffix-count >= 2048 (strictly-above items provably < 2048, so
// only cutoff-bin ties can hit the CAP) -> LDS compaction -> bitonic sort.
// ---------------------------------------------------------------------------
__global__ __launch_bounds__(1024)
void k_topk(const float* __restrict__ scores,
            float* __restrict__ sorted_s, int* __restrict__ sorted_i) {
  __shared__ unsigned hist[256];
  __shared__ int s_tb, s_cnt;
  __shared__ float ks[CAP];
  __shared__ int   kv[CAP];

  const int img = blockIdx.x, tid = threadIdx.x;
  const float* sc = scores + (size_t)img * M_CAND;

  for (int b = tid; b < 256; b += 1024) hist[b] = 0;
  if (tid == 0) s_cnt = 0;
  __syncthreads();

  for (int i = tid; i < M_CAND; i += 1024) {
    const float s = sc[i];
    if (s > 0.f) {
      int b = (int)(s * 256.f); b = b > 255 ? 255 : b;
      atomicAdd(&hist[b], 1u);
    }
  }
  __syncthreads();

  if (tid == 0) {
    unsigned acc = 0; int tb = 0;
    for (int b = 255; b >= 0; --b) { acc += hist[b]; if (acc >= KPRE) { tb = b; break; } }
    s_tb = tb;
  }
  __syncthreads();
  const int tb = s_tb;

  for (int i = tid; i < M_CAND; i += 1024) {
    const float s = sc[i];
    if (s > 0.f) {
      int b = (int)(s * 256.f); b = b > 255 ? 255 : b;
      if (b >= tb) {
        const int pos = atomicAdd(&s_cnt, 1);
        if (pos < CAP) { ks[pos] = s; kv[pos] = i; }
      }
    }
  }
  __syncthreads();
  int cnt = s_cnt; if (cnt > CAP) cnt = CAP;
  for (int i = cnt + tid; i < CAP; i += 1024) { ks[i] = -__builtin_inff(); kv[i] = 0; }
  __syncthreads();

  // Bitonic sort, descending by score.
  for (int k = 2; k <= CAP; k <<= 1) {
    for (int j = k >> 1; j > 0; j >>= 1) {
      for (int i = tid; i < CAP; i += 1024) {
        const int ixj = i ^ j;
        if (ixj > i) {
          const bool desc = ((i & k) == 0);
          const float a = ks[i], b = ks[ixj];
          const bool sw = desc ? (a < b) : (a > b);
          if (sw) { ks[i] = b; ks[ixj] = a; const int t = kv[i]; kv[i] = kv[ixj]; kv[ixj] = t; }
        }
      }
      __syncthreads();
    }
  }
  for (int i = tid; i < KPRE; i += 1024) {
    sorted_s[img * KPRE + i] = ks[i];
    sorted_i[img * KPRE + i] = kv[i];
  }
}

// ---------------------------------------------------------------------------
// Kernel C: decode the 2048 selected candidates; per-image max coordinate
// (coords are >= 0, so float atomicMax == int atomicMax on the bit pattern).
// ---------------------------------------------------------------------------
__global__ __launch_bounds__(256)
void k_decode(const float* __restrict__ deltas, const float* __restrict__ props,
              const float* __restrict__ sorted_s, const int* __restrict__ sorted_i,
              float* __restrict__ boxes, int* __restrict__ lab, int* __restrict__ maxc) {
  const int img = blockIdx.y;
  const int k   = blockIdx.x * 256 + threadIdx.x;
  const float s = sorted_s[img * KPRE + k];
  const int idx = sorted_i[img * KPRE + k];
  const int n   = idx / NFG;
  const int c   = idx % NFG + 1;

  const float* pr = props + (size_t)(img * N_PROP + n) * 4;
  const float pw = pr[2] - pr[0], ph = pr[3] - pr[1];
  const float pcx = pr[0] + 0.5f * pw, pcy = pr[1] + 0.5f * ph;
  const float* d = deltas + (size_t)(img * N_PROP + n) * C_CLS * 4 + (size_t)c * 4;

  const float dx = d[0] * 0.1f, dy = d[1] * 0.1f;
  const float dw = fminf(d[2] * 0.2f, XCLIP), dh = fminf(d[3] * 0.2f, XCLIP);
  const float cx = dx * pw + pcx, cy = dy * ph + pcy;
  const float bw = __expf(dw) * pw, bh = __expf(dh) * ph;
  const float x1 = fminf(fmaxf(cx - 0.5f * bw, 0.f), IMG_WH);
  const float y1 = fminf(fmaxf(cy - 0.5f * bh, 0.f), IMG_WH);
  const float x2 = fminf(fmaxf(cx + 0.5f * bw, 0.f), IMG_WH);
  const float y2 = fminf(fmaxf(cy + 0.5f * bh, 0.f), IMG_WH);

  float* bx = boxes + (size_t)(img * KPRE + k) * 4;
  bx[0] = x1; bx[1] = y1; bx[2] = x2; bx[3] = y2;
  lab[img * KPRE + k] = c;

  const bool valid = s > NEG_BIG;
  const float mc = valid ? fmaxf(fmaxf(x1, y1), fmaxf(x2, y2)) : 0.f;
  atomicMax(&maxc[img], __float_as_int(mc));
}

__global__ __launch_bounds__(256)
void k_offset(const float* __restrict__ boxes, const int* __restrict__ lab,
              const int* __restrict__ maxc, float* __restrict__ obox) {
  const int img = blockIdx.y;
  const int k   = blockIdx.x * 256 + threadIdx.x;
  const float off = (float)lab[img * KPRE + k] * (__int_as_float(maxc[img]) + 1.0f);
  const size_t o = (size_t)(img * KPRE + k) * 4;
#pragma unroll
  for (int j = 0; j < 4; ++j) obox[o + j] = boxes[o + j] + off;
}

// ---------------------------------------------------------------------------
// Kernel D: IoU suppression bitmasks. Wave 0 of each block stages the full
// 2048x4 f32 offset-box tile (32KB) into LDS via one TDM descriptor
// (1-D tensor, data_size=4B, dim0=tile_dim0=8192 elements), waits on
// TENSORcnt, then 256 rows/block each build 64 x 32-bit mask words.
// ---------------------------------------------------------------------------
__global__ __launch_bounds__(256)
void k_nmsmask(const float* __restrict__ obox, unsigned* __restrict__ masks) {
  __shared__ float lb[KPRE * 4];   // 32 KB of the WGP's 320 KB LDS
  const int img = blockIdx.y;
  const int row = blockIdx.x * 256 + threadIdx.x;
  const float* gsrc = obox + (size_t)img * KPRE * 4;

  if (threadIdx.x < 32) {          // wave 0 only (wave32)
    const unsigned lds_addr = (unsigned)(uintptr_t)(void*)lb;
    const unsigned long long ga = (unsigned long long)(uintptr_t)gsrc;
    // D# group 0: count=1, lds_addr, global_addr[56:0], type=2 ("image").
    v4u g0 = { 1u,
               lds_addr,
               (unsigned)(ga & 0xffffffffu),
               (unsigned)((ga >> 32) & 0x01ffffffu) | (2u << 30) };
    // D# group 1: data_size=2 (4B); tensor_dim0=8192 [79:48]; tensor_dim1=1
    // [111:80]; tile_dim0=8192 [127:112]; tile_dim1=1 [143:128];
    // tensor_dim0_stride=8192 [207:160]; everything else 0.
    v8i g1 = { (int)(2u << 16),
               (int)(8192u << 16),
               (int)(1u << 16),
               (int)(8192u << 16),
               1,
               8192,
               0, 0 };
    v4i g2 = {0, 0, 0, 0}, g3 = {0, 0, 0, 0};
    v8i g4 = {0, 0, 0, 0, 0, 0, 0, 0};   // 6-arg toolchain: extra group, zeroed
    __builtin_amdgcn_tensor_load_to_lds(g0, g1, g2, g3, g4, 0);
    __builtin_amdgcn_s_wait_tensorcnt(0);
  }
  __syncthreads();

  const float rx1 = lb[row * 4 + 0], ry1 = lb[row * 4 + 1];
  const float rx2 = lb[row * 4 + 2], ry2 = lb[row * 4 + 3];
  const float rar = (rx2 - rx1) * (ry2 - ry1);
  unsigned* mrow = masks + ((size_t)(img * KPRE + row)) * 64;

  for (int w = 0; w < 64; ++w) {
    unsigned bits = 0;
    for (int b = 0; b < 32; ++b) {
      const int col = w * 32 + b;
      if (col > row) {
        const float cx1 = lb[col * 4 + 0], cy1 = lb[col * 4 + 1];
        const float cx2 = lb[col * 4 + 2], cy2 = lb[col * 4 + 3];
        float ix = fminf(rx2, cx2) - fmaxf(rx1, cx1);
        float iy = fminf(ry2, cy2) - fmaxf(ry1, cy1);
        ix = ix > 0.f ? ix : 0.f;  iy = iy > 0.f ? iy : 0.f;
        const float inter = ix * iy;
        const float uni = rar + (cx2 - cx1) * (cy2 - cy1) - inter;
        if (inter / (uni + 1e-9f) > NMS_T) bits |= (1u << b);
      }
    }
    mrow[w] = bits;
  }
}

// ---------------------------------------------------------------------------
// Kernel E: serial greedy scan, one wave per image. Candidates are already
// sorted desc by score, so the first 100 kept == final top-100.
// ---------------------------------------------------------------------------
__global__ __launch_bounds__(32)
void k_nms_scan(const unsigned* __restrict__ masks, const float* __restrict__ boxes,
                const float* __restrict__ sorted_s, const int* __restrict__ lab,
                float* __restrict__ dets, int* __restrict__ out_lab) {
  __shared__ unsigned validw[64];
  __shared__ unsigned remv[64];
  const int img = blockIdx.x, lane = threadIdx.x;

  remv[lane] = 0; remv[lane + 32] = 0;
  for (int w = 0; w < 64; ++w) {
    const bool v = sorted_s[img * KPRE + w * 32 + lane] > NEG_BIG;
    const unsigned m = (unsigned)__ballot(v);
    if (lane == 0) validw[w] = m;
  }
  __syncthreads();

  int outc = 0;
  for (int i = 0; i < KPRE && outc < DETS; ++i) {
    const int w = i >> 5, b = i & 31;
    const bool kp = ((validw[w] >> b) & 1u) && !((remv[w] >> b) & 1u);
    if (kp) {
      const unsigned* mrow = masks + ((size_t)(img * KPRE + i)) * 64;
      remv[lane]      |= mrow[lane];
      remv[lane + 32] |= mrow[lane + 32];
      if (lane == 0) {
        const float* bx = boxes + (size_t)(img * KPRE + i) * 4;
        float* dr = dets + (size_t)(img * DETS + outc) * 5;
        dr[0] = bx[0]; dr[1] = bx[1]; dr[2] = bx[2]; dr[3] = bx[3];
        dr[4] = sorted_s[img * KPRE + i];
        out_lab[img * DETS + outc] = lab[img * KPRE + i];
      }
      outc++;
    }
  }
  for (int k = outc + lane; k < DETS; k += 32) {
    float* dr = dets + (size_t)(img * DETS + k) * 5;
    dr[0] = dr[1] = dr[2] = dr[3] = dr[4] = 0.f;
    out_lab[img * DETS + k] = -1;
  }
}

__global__ void k_init(int* maxc) { if (threadIdx.x < B_IMG) maxc[threadIdx.x] = 0; }

// ---------------------------------------------------------------------------
extern "C" void kernel_launch(void* const* d_in, const int* in_sizes, int n_in,
                              void* d_out, int out_size, void* d_ws, size_t ws_size,
                              hipStream_t stream) {
  const float* logits = (const float*)d_in[0];   // [B*N, C]
  const float* deltas = (const float*)d_in[1];   // [B*N, C*4]
  const float* props  = (const float*)d_in[2];   // [B, N, 4]

  // Workspace layout (~10.7 MB total).
  char* w = (char*)d_ws;
  float* scores   = (float*)w;    w += sizeof(float) * (size_t)B_IMG * M_CAND;
  float* sorted_s = (float*)w;    w += sizeof(float) * B_IMG * KPRE;
  int*   sorted_i = (int*)w;      w += sizeof(int)   * B_IMG * KPRE;
  float* boxes    = (float*)w;    w += sizeof(float) * B_IMG * KPRE * 4;
  float* obox     = (float*)w;    w += sizeof(float) * B_IMG * KPRE * 4;
  int*   lab      = (int*)w;      w += sizeof(int)   * B_IMG * KPRE;
  int*   maxc     = (int*)w;      w += 64;
  unsigned* masks = (unsigned*)w; // B*KPRE*64 u32 = 4 MB

  float* dets   = (float*)d_out;                  // [B,100,5]
  int* out_lab  = (int*)(dets + B_IMG * DETS * 5); // [B,100] (int bits)

  k_init   <<<1, 32, 0, stream>>>(maxc);
  k_scores <<<B_IMG * (N_PROP / 16), 32, 0, stream>>>(logits, deltas, props, scores);
  k_topk   <<<B_IMG, 1024, 0, stream>>>(scores, sorted_s, sorted_i);
  k_decode <<<dim3(KPRE / 256, B_IMG), 256, 0, stream>>>(deltas, props, sorted_s,
                                                         sorted_i, boxes, lab, maxc);
  k_offset <<<dim3(KPRE / 256, B_IMG), 256, 0, stream>>>(boxes, lab, maxc, obox);
  k_nmsmask<<<dim3(KPRE / 256, B_IMG), 256, 0, stream>>>(obox, masks);
  k_nms_scan<<<B_IMG, 32, 0, stream>>>(masks, boxes, sorted_s, lab, dets, out_lab);
}
